// BigramLanguageModel_463856467968
// MI455X (gfx1250) — matrix operand order repaired and checked by reading, not compiled, not run on hardware
//
#include <hip/hip_runtime.h>
#include <hip/hip_bf16.h>

typedef __attribute__((ext_vector_type(16))) __bf16 v16bf;
typedef __attribute__((ext_vector_type(2)))  __bf16 v2bf;
typedef __attribute__((ext_vector_type(8)))  float  v8f;

union BF16x16 { v16bf v; unsigned u[8]; unsigned short s[16]; };

// Native packed f32->bf16 conversion (v_cvt_pk_bf16_f32) when available.
// a -> low 16 bits, b -> high 16 bits.
__device__ inline unsigned pk2(float a, float b) {
#if __has_builtin(__builtin_amdgcn_cvt_pk_bf16_f32)
  union { v2bf v; unsigned u; } c;
  c.v = __builtin_amdgcn_cvt_pk_bf16_f32(a, b);
  return c.u;
#else
  union { __bf16 h[2]; unsigned u; } c;
  c.h[0] = (__bf16)a; c.h[1] = (__bf16)b;
  return c.u;
#endif
}

// ---------------------------------------------------------------------------
// Embedding: x[row] = tok_emb[idx[row]] + pos_emb[row % T]
// ---------------------------------------------------------------------------
__global__ __launch_bounds__(256)
void embed_kernel(const int* __restrict__ idx, const float* __restrict__ tok,
                  const float* __restrict__ pos, float* __restrict__ x,
                  int T, int D) {
  int row = blockIdx.x;
  int t = row % T;
  int token = idx[row];
  for (int d = threadIdx.x; d < D; d += blockDim.x)
    x[(size_t)row * D + d] = tok[(size_t)token * D + d] + pos[(size_t)t * D + d];
}

// ---------------------------------------------------------------------------
// LayerNorm: one workgroup per row
// ---------------------------------------------------------------------------
__global__ __launch_bounds__(256)
void ln_kernel(const float* __restrict__ x, const float* __restrict__ g,
               const float* __restrict__ b, float* __restrict__ out, int D) {
  __shared__ float r1[256], r2[256];
  int row = blockIdx.x, tid = threadIdx.x;
  const float* xr = x + (size_t)row * D;
  float s = 0.f, s2 = 0.f;
  for (int i = tid; i < D; i += 256) { float v = xr[i]; s += v; s2 += v * v; }
  r1[tid] = s; r2[tid] = s2; __syncthreads();
  for (int st = 128; st > 0; st >>= 1) {
    if (tid < st) { r1[tid] += r1[tid + st]; r2[tid] += r2[tid + st]; }
    __syncthreads();
  }
  float mean = r1[0] / (float)D;
  float var  = r2[0] / (float)D - mean * mean;
  float rstd = rsqrtf(var + 1e-5f);
  for (int i = tid; i < D; i += 256)
    out[(size_t)row * D + i] = (xr[i] - mean) * rstd * g[i] + b[i];
}

// ---------------------------------------------------------------------------
// Tile staging: fp32 global -> bf16 LDS (A row-major [row][k], W transposed
// [n][k]).  N may be ragged (50257); A requires M%128==0, K%32==0.
// ---------------------------------------------------------------------------
__device__ __forceinline__
void stage_tiles(const float* __restrict__ A, const float* __restrict__ W,
                 int N, int K, int m0, int n0, int k0, int tid,
                 unsigned short* __restrict__ As16,
                 unsigned short* __restrict__ Bs16) {
#pragma unroll
  for (int it = 0; it < 4; ++it) {
    int idx4 = tid * 4 + it;                 // 1024 float4 loads
    int r  = idx4 >> 3;                      // 0..127
    int c4 = (idx4 & 7) << 2;                // 0..28
    const float* ap = A + (size_t)(m0 + r) * K + (k0 + c4);
    float4 f = *(const float4*)ap;
    *(unsigned*)&As16[r * 32 + c4]     = pk2(f.x, f.y);
    *(unsigned*)&As16[r * 32 + c4 + 2] = pk2(f.z, f.w);
  }
  // W: each thread packs a k-pair for one n column (lane-consecutive n)
#pragma unroll
  for (int it = 0; it < 8; ++it) {
    int idx = tid + it * 256;                // 0..2047
    int kp  = idx >> 7;                      // k pair 0..15
    int n   = idx & 127;
    int gn  = n0 + n;
    float w0 = 0.f, w1 = 0.f;
    if (gn < N) {
      size_t base = (size_t)(k0 + 2 * kp) * N + gn;
      w0 = W[base];
      w1 = W[base + N];
    }
    *(unsigned*)&Bs16[n * 32 + 2 * kp] = pk2(w0, w1);
  }
}

// ---------------------------------------------------------------------------
// GEMM: C[M,N] = A[M,K] @ W[K,N] (+bias) (+residual) (+ReLU)
// Grid: x = M blocks, y = N blocks -> consecutive WGs share one W column
// panel through L2, so fp32 weights stream from HBM exactly once (critical
// for the 206MB LM head).  Double-buffered LDS staging, L2 prefetch of the
// k0+64 tile, math on v_wmma_f32_16x16x32_bf16.
// Block 256 thr = 8 waves; block tile 128x128, BK=32; wave tile 32x64.
// ---------------------------------------------------------------------------
__global__ __launch_bounds__(256)
void gemm_bf16_wmma(const float* __restrict__ A, const float* __restrict__ W,
                    const float* __restrict__ bias, const float* __restrict__ resid,
                    float* __restrict__ C, int M, int N, int K, int relu) {
  __shared__ unsigned short As16[2][128 * 32];   // [row][k]      2 x 8 KB
  __shared__ unsigned short Bs16[2][128 * 32];   // [n][k] (W^T)  2 x 8 KB

  const int tid  = threadIdx.x;
  const int lane = tid & 31, wave = tid >> 5;
  const int hf = lane >> 4, lq = lane & 15;
  const int wm = wave & 3, wn = wave >> 2;       // 4 waves in M, 2 in N
  const int m0 = blockIdx.x * 128, n0 = blockIdx.y * 128;
  const int wr0 = wm * 32, wc0 = wn * 64;

  v8f acc[2][4];
#pragma unroll
  for (int i = 0; i < 2; i++)
#pragma unroll
    for (int j = 0; j < 4; j++)
#pragma unroll
      for (int e = 0; e < 8; e++) acc[i][j][e] = 0.f;

  stage_tiles(A, W, N, K, m0, n0, 0, tid, As16[0], Bs16[0]);

  const int nk = K >> 5;
  for (int kb = 0; kb < nk; ++kb) {
    const int p = kb & 1;
    __syncthreads();                             // stage(p) complete, reads(p^1) done

    const unsigned* AsU = (const unsigned*)As16[p];
    const unsigned* BsU = (const unsigned*)Bs16[p];
    BF16x16 afr[2], bfr[4];
#pragma unroll
    for (int i = 0; i < 2; i++) {
      int row = wr0 + i * 16 + lq;
#pragma unroll
      for (int q = 0; q < 4; q++) {
        afr[i].u[q]     = AsU[row * 16 + hf * 4 + q];
        afr[i].u[4 + q] = AsU[row * 16 + 8 + hf * 4 + q];
      }
    }
#pragma unroll
    for (int j = 0; j < 4; j++) {
      int col = wc0 + j * 16 + lq;
#pragma unroll
      for (int q = 0; q < 8; q++) bfr[j].u[q] = BsU[col * 16 + (q >> 2) * 8 + hf * 4 + (q & 3)];
    }

    if (kb + 1 < nk)                             // stage next tile into p^1
      stage_tiles(A, W, N, K, m0, n0, (kb + 1) * 32, tid, As16[p ^ 1], Bs16[p ^ 1]);

    if (kb + 2 < nk) {                           // L2 prefetch of tile after next
      int k2 = (kb + 2) * 32;
      int r  = (tid * 4) >> 3;
      int c4 = ((tid * 4) & 7) << 2;
      __builtin_prefetch(A + (size_t)(m0 + r) * K + (k2 + c4), 0, 3);
      int kk = tid >> 5;
      int n4 = (tid & 31) << 2;
      __builtin_prefetch(W + (size_t)(k2 + kk) * N + (size_t)(n0 + n4), 0, 3);
    }

#pragma unroll
    for (int i = 0; i < 2; i++)
#pragma unroll
      for (int j = 0; j < 4; j++)
        acc[i][j] = __builtin_amdgcn_wmma_f32_16x16x32_bf16(
            false, afr[i].v, false, bfr[j].v, (short)0, acc[i][j], false, false);
  }

  // ---- epilogue: bias / residual / relu, C layout row = r + hf*8, col = lq
#pragma unroll
  for (int j = 0; j < 4; j++) {
    int gcol = n0 + wc0 + j * 16 + lq;
    if (gcol >= N) continue;
    float bv = bias ? bias[gcol] : 0.f;
#pragma unroll
    for (int i = 0; i < 2; i++) {
#pragma unroll
      for (int r = 0; r < 8; r++) {
        int grow = m0 + wr0 + i * 16 + hf * 8 + r;
        size_t o = (size_t)grow * N + gcol;
        float v = acc[i][j][r] + bv;
        if (resid) v += resid[o];
        if (relu) v = fmaxf(v, 0.f);
        C[o] = v;
      }
    }
  }
}

// ---------------------------------------------------------------------------
// Flash attention: 4 waves per WG cover a 64-query chunk of one (batch,head)
// and SHARE the K/V LDS staging (4x less K/V traffic).  Each wave owns a
// 16-query tile: S = Q K^T via 4 WMMA / 32-key block; online softmax
// (width-16 shuffle reductions); P restaged via per-wave LDS into A-fragment
// layout; O += P V via 4 WMMA / block.  Uniform block count across waves:
// fully-masked blocks yield P=0 and are harmless.
// ---------------------------------------------------------------------------
__global__ __launch_bounds__(128)
void attn_kernel(const float* __restrict__ Q, const float* __restrict__ Kb,
                 const float* __restrict__ Vb, float* __restrict__ O,
                 int T, int D, int DH, float scale) {
  __shared__ unsigned short Ks[32 * 64];      // [key][d] bf16, 4 KB
  __shared__ unsigned short Vs[32 * 64];      // [key][d] bf16, 4 KB
  __shared__ unsigned short Ps[4][16 * 32];   // per-wave [m][s] bf16, 4 KB
  const unsigned* KsU = (const unsigned*)Ks;

  const int tid  = threadIdx.x;
  const int wv   = tid >> 5;
  const int lane = tid & 31;
  const int hf = lane >> 4, lq = lane & 15;
  const int qc = blockIdx.x, hh = blockIdx.y, bb = blockIdx.z;
  const int t0 = qc * 64 + wv * 16;

  // Q fragments (A layout), pre-scaled
  const float* qrow = Q + (size_t)(bb * T + t0 + lq) * D + hh * DH;
  BF16x16 qa[2];
#pragma unroll
  for (int db = 0; db < 2; db++) {
#pragma unroll
    for (int p = 0; p < 4; p++) {
      float2 f0 = *(const float2*)(qrow + db * 32 + hf * 8 + 2 * p);
      float2 f1 = *(const float2*)(qrow + db * 32 + 16 + hf * 8 + 2 * p);
      qa[db].u[p]     = pk2(f0.x * scale, f0.y * scale);
      qa[db].u[4 + p] = pk2(f1.x * scale, f1.y * scale);
    }
  }

  float mrow[8], lrow[8];
  v8f accO[4];
#pragma unroll
  for (int r = 0; r < 8; r++) { mrow[r] = -3e30f; lrow[r] = 0.f; }
#pragma unroll
  for (int dt = 0; dt < 4; dt++)
#pragma unroll
    for (int e = 0; e < 8; e++) accO[dt][e] = 0.f;

  const int nblk = 2 * (qc + 1);              // keys <= qc*64+63
  for (int blk = 0; blk < nblk; ++blk) {
    int s0 = blk * 32;
    // cooperative staging: 128 threads, thread -> (key, 16-float segment)
    {
      int key = tid >> 2;
      int seg = (tid & 3) * 16;
      const float* krow = Kb + (size_t)(bb * T + s0 + key) * D + hh * DH + seg;
      const float* vrow = Vb + (size_t)(bb * T + s0 + key) * D + hh * DH + seg;
#pragma unroll
      for (int i = 0; i < 4; i++) {
        float4 fk = *(const float4*)(krow + i * 4);
        float4 fv = *(const float4*)(vrow + i * 4);
        *(unsigned*)&Ks[key * 64 + seg + i * 4]     = pk2(fk.x, fk.y);
        *(unsigned*)&Ks[key * 64 + seg + i * 4 + 2] = pk2(fk.z, fk.w);
        *(unsigned*)&Vs[key * 64 + seg + i * 4]     = pk2(fv.x, fv.y);
        *(unsigned*)&Vs[key * 64 + seg + i * 4 + 2] = pk2(fv.z, fv.w);
      }
    }
    __syncthreads();

    // S = Q K^T : two 16x16 score tiles (keys s0..+15, s0+16..+31)
    v8f sc[2];
#pragma unroll
    for (int kt = 0; kt < 2; kt++) {
#pragma unroll
      for (int e = 0; e < 8; e++) sc[kt][e] = 0.f;
      int key = kt * 16 + lq;
#pragma unroll
      for (int db = 0; db < 2; db++) {
        BF16x16 kf;
#pragma unroll
        for (int p = 0; p < 8; p++) kf.u[p] = KsU[key * 32 + db * 16 + (p >> 2) * 8 + hf * 4 + (p & 3)];
        sc[kt] = __builtin_amdgcn_wmma_f32_16x16x32_bf16(
            false, qa[db].v, false, kf.v, (short)0, sc[kt], false, false);
      }
    }

    // online softmax (rows r live at tile row hf*8+r; cols across 16 lanes)
#pragma unroll
    for (int r = 0; r < 8; r++) {
      int rowt = t0 + hf * 8 + r;
      float a0 = sc[0][r]; if (s0 + lq > rowt)      a0 = -3e30f;
      float a1 = sc[1][r]; if (s0 + 16 + lq > rowt) a1 = -3e30f;
      float mx = fmaxf(a0, a1);
#pragma unroll
      for (int d = 1; d < 16; d <<= 1) mx = fmaxf(mx, __shfl_xor(mx, d, 16));
      float mnew = fmaxf(mrow[r], mx);
      float p0 = __expf(a0 - mnew);
      float p1 = __expf(a1 - mnew);
      float rs = p0 + p1;
#pragma unroll
      for (int d = 1; d < 16; d <<= 1) rs += __shfl_xor(rs, d, 16);
      float corr = __expf(mrow[r] - mnew);
      lrow[r] = lrow[r] * corr + rs;
      mrow[r] = mnew;
#pragma unroll
      for (int dt = 0; dt < 4; dt++) accO[dt][r] *= corr;
      unsigned pp = pk2(p0, p1);
      Ps[wv][(hf * 8 + r) * 32 + lq]      = (unsigned short)pp;
      Ps[wv][(hf * 8 + r) * 32 + 16 + lq] = (unsigned short)(pp >> 16);
    }
    __syncthreads();

    // O += P V : P as A-fragment, V as B-fragment (scalar 16-bit gathers)
    const unsigned* PsU = (const unsigned*)Ps[wv];
    BF16x16 pf;
#pragma unroll
    for (int p = 0; p < 4; p++) {
      pf.u[p]     = PsU[lq * 16 + hf * 4 + p];
      pf.u[4 + p] = PsU[lq * 16 + 8 + hf * 4 + p];
    }
#pragma unroll
    for (int dt = 0; dt < 4; dt++) {
      BF16x16 vf;
#pragma unroll
      for (int j = 0; j < 16; j++)
        vf.s[j] = Vs[((j >> 3) * 16 + hf * 8 + (j & 7)) * 64 + dt * 16 + lq];
      accO[dt] = __builtin_amdgcn_wmma_f32_16x16x32_bf16(
          false, pf.v, false, vf.v, (short)0, accO[dt], false, false);
    }
    __syncthreads();                            // Ks/Vs reads done before restage
  }

  // normalize and write
#pragma unroll
  for (int dt = 0; dt < 4; dt++) {
#pragma unroll
    for (int r = 0; r < 8; r++) {
      int rowt = t0 + hf * 8 + r;
      float inv = 1.0f / lrow[r];
      O[(size_t)(bb * T + rowt) * D + hh * DH + dt * 16 + lq] = accO[dt][r] * inv;
    }
  }
}

// ---------------------------------------------------------------------------
// Per-row log-softmax loss (two passes; second pass hits L2: one row=200KB)
// ---------------------------------------------------------------------------
__global__ __launch_bounds__(256)
void rowloss_kernel(const float* __restrict__ logits, const int* __restrict__ targets,
                    float* __restrict__ rl, int V) {
  __shared__ float red[256];
  int row = blockIdx.x, tid = threadIdx.x;
  const float* lg = logits + (size_t)row * V;
  float mx = -3.4e38f;
  for (int i = tid; i < V; i += 256) mx = fmaxf(mx, lg[i]);
  red[tid] = mx; __syncthreads();
  for (int s = 128; s > 0; s >>= 1) {
    if (tid < s) red[tid] = fmaxf(red[tid], red[tid + s]);
    __syncthreads();
  }
  mx = red[0]; __syncthreads();
  float sum = 0.f;
  for (int i = tid; i < V; i += 256) sum += __expf(lg[i] - mx);
  red[tid] = sum; __syncthreads();
  for (int s = 128; s > 0; s >>= 1) {
    if (tid < s) red[tid] += red[tid + s];
    __syncthreads();
  }
  if (tid == 0) {
    int tgt = targets[row];
    rl[row] = -(lg[tgt] - mx - logf(red[0]));
  }
}

__global__ __launch_bounds__(256)
void loss_reduce_kernel(const float* __restrict__ rl, float* __restrict__ out, int n) {
  __shared__ float red[256];
  int tid = threadIdx.x;
  float s = 0.f;
  for (int i = tid; i < n; i += 256) s += rl[i];
  red[tid] = s; __syncthreads();
  for (int st = 128; st > 0; st >>= 1) {
    if (tid < st) red[tid] += red[tid + st];
    __syncthreads();
  }
  if (tid == 0) *out = red[0] / (float)n;
}

// ---------------------------------------------------------------------------
extern "C" void kernel_launch(void* const* d_in, const int* in_sizes, int n_in,
                              void* d_out, int out_size, void* d_ws, size_t ws_size,
                              hipStream_t stream) {
  (void)in_sizes; (void)n_in; (void)out_size; (void)ws_size;

  const int Bn = 4, T = 1024, Dm = 1024, Ln = 8, Hh = 16, DHd = 64, Vn = 50257;
  const int Mr = Bn * T;                    // 4096 rows

  const int*   idx     = (const int*)d_in[0];
  const int*   targets = (const int*)d_in[1];
  const float* tok_emb = (const float*)d_in[2];
  const float* pos_emb = (const float*)d_in[3];
  const float* Wq   = (const float*)d_in[4];
  const float* Wk   = (const float*)d_in[5];
  const float* Wv   = (const float*)d_in[6];
  const float* Wo   = (const float*)d_in[7];
  const float* bo   = (const float*)d_in[8];
  const float* ln1g = (const float*)d_in[9];
  const float* ln1b = (const float*)d_in[10];
  const float* ln2g = (const float*)d_in[11];
  const float* ln2b = (const float*)d_in[12];
  const float* W1   = (const float*)d_in[13];
  const float* b1   = (const float*)d_in[14];
  const float* W2   = (const float*)d_in[15];
  const float* b2   = (const float*)d_in[16];
  const float* lnfg = (const float*)d_in[17];
  const float* lnfb = (const float*)d_in[18];
  const float* Wlm  = (const float*)d_in[19];
  const float* blm  = (const float*)d_in[20];

  float* ws = (float*)d_ws;
  size_t off = 0;
  float* x   = ws + off; off += (size_t)Mr * Dm;
  float* h   = ws + off; off += (size_t)Mr * Dm;
  float* qb  = ws + off; off += (size_t)Mr * Dm;
  float* kb  = ws + off; off += (size_t)Mr * Dm;
  float* vb  = ws + off; off += (size_t)Mr * Dm;
  float* att = ws + off; off += (size_t)Mr * Dm;
  float* hid = ws + off; off += (size_t)Mr * 4 * Dm;
  float* rl  = ws + off; off += (size_t)Mr;

  float* logits = (float*)d_out;
  float* loss   = logits + (size_t)Mr * Vn;

  const float scale = 0.03125f;             // D^-0.5 = 1024^-0.5

  embed_kernel<<<Mr, 256, 0, stream>>>(idx, tok_emb, pos_emb, x, T, Dm);

  dim3 gD(Mr / 128, Dm / 128);              // (32, 8)   x=M, y=N
  dim3 g4D(Mr / 128, 4 * Dm / 128);         // (32, 32)
  dim3 gV(Mr / 128, (Vn + 127) / 128);      // (32, 393)
  dim3 gA(T / 64, Hh, Bn);                  // (16, 16, 4)

  for (int l = 0; l < Ln; ++l) {
    const float* Wq_l = Wq + (size_t)l * Dm * Dm;
    const float* Wk_l = Wk + (size_t)l * Dm * Dm;
    const float* Wv_l = Wv + (size_t)l * Dm * Dm;
    const float* Wo_l = Wo + (size_t)l * Dm * Dm;
    const float* W1_l = W1 + (size_t)l * Dm * 4 * Dm;
    const float* W2_l = W2 + (size_t)l * 4 * Dm * Dm;

    ln_kernel<<<Mr, 256, 0, stream>>>(x, ln1g + (size_t)l * Dm, ln1b + (size_t)l * Dm, h, Dm);
    gemm_bf16_wmma<<<gD, 256, 0, stream>>>(h, Wq_l, nullptr, nullptr, qb, Mr, Dm, Dm, 0);
    gemm_bf16_wmma<<<gD, 256, 0, stream>>>(h, Wk_l, nullptr, nullptr, kb, Mr, Dm, Dm, 0);
    gemm_bf16_wmma<<<gD, 256, 0, stream>>>(h, Wv_l, nullptr, nullptr, vb, Mr, Dm, Dm, 0);
    attn_kernel<<<gA, 128, 0, stream>>>(qb, kb, vb, att, T, Dm, DHd, scale);
    gemm_bf16_wmma<<<gD, 256, 0, stream>>>(att, Wo_l, bo + (size_t)l * Dm, x, x, Mr, Dm, Dm, 0);
    ln_kernel<<<Mr, 256, 0, stream>>>(x, ln2g + (size_t)l * Dm, ln2b + (size_t)l * Dm, h, Dm);
    gemm_bf16_wmma<<<g4D, 256, 0, stream>>>(h, W1_l, b1 + (size_t)l * 4 * Dm, nullptr, hid, Mr, 4 * Dm, Dm, 1);
    gemm_bf16_wmma<<<gD, 256, 0, stream>>>(hid, W2_l, b2 + (size_t)l * Dm, x, x, Mr, Dm, 4 * Dm, 0);
  }

  ln_kernel<<<Mr, 256, 0, stream>>>(x, lnfg, lnfb, h, Dm);
  gemm_bf16_wmma<<<gV, 256, 0, stream>>>(h, Wlm, blm, nullptr, logits, Mr, Vn, Dm, 0);
  rowloss_kernel<<<Mr, 256, 0, stream>>>(logits, targets, rl, Vn);
  loss_reduce_kernel<<<1, 256, 0, stream>>>(rl, loss, Mr);
}
